// GlobalRpeDecompAttentionWithPositionEmbedding_80255758893719
// MI455X (gfx1250) — matrix operand order repaired and checked by reading, not compiled, not run on hardware
//
#include <hip/hip_runtime.h>
#include <math.h>

// ---------------------------------------------------------------------------
// MI455X (gfx1250) multi-head attention, bf16 WMMA + async global->LDS staging.
//   B=2, NQ=1024, S=8192, D=256, H=8, hd=32  -> hd==32 == WMMA K for bf16.
// Pipeline: Q/K/V projections (WMMA, bf16 out, attention-friendly layouts)
//           -> flash-attention: 8-wave workgroup shares one (b,h); K/V staged
//              into LDS with GLOBAL_LOAD_ASYNC_TO_LDS_B128 (double-buffered,
//              ASYNCcnt-synchronized); 8 WMMAs per 64-key chunk per wave
//           -> output projection (WMMA, f32 out).
// Workspace: qbf 1MB | kbf 8MB | vbfT 8MB | ctxf 2MB  (~19MB of d_ws).
// ---------------------------------------------------------------------------

typedef __attribute__((ext_vector_type(16))) __bf16 v16bf;
typedef __attribute__((ext_vector_type(8)))  float  v8f;

#define DIMD  256
#define NHEAD 8
#define HD    32
#define NQTOK 1024
#define SEQ   8192
#define BATCH 2
#define CHUNK 64                 // keys staged per workgroup iteration

__device__ __forceinline__ v8f wmma_bf16(v16bf a, v16bf b, v8f c) {
  // (neg_a, A, neg_b, B, c_mod, C, reuse_a, reuse_b)
  return __builtin_amdgcn_wmma_f32_16x16x32_bf16(false, a, false, b, (short)0, c,
                                                 false, false);
}

// 16-byte async copy global -> LDS (tracked by ASYNCcnt).
// VDST = VGPR holding the LDS byte address, VADDR = 64-bit global address.
__device__ __forceinline__ void async_copy_b128(const __bf16* g, __bf16* l) {
  unsigned           lds = (unsigned)(uintptr_t)l;     // low 32b = LDS offset
  unsigned long long ga  = (unsigned long long)(uintptr_t)g;
  asm volatile("global_load_async_to_lds_b128 %0, %1, off"
               :: "v"(lds), "v"(ga) : "memory");
}

// ---------------------------------------------------------------------------
// Generic 16x16-tile GEMM:  Y = X(rows x 256) @ W^T(256 x 256) + bias
// One wave per output tile, 8 waves / 256-thread block.
// mode 0: Q -> bf16, *scale, (B,H,pos,hd) | 1: K -> bf16 (B,H,pos,hd)
// mode 2: V -> bf16 transposed (B,H,hd,pos) | 3: O -> f32 row-major (d_out)
// ---------------------------------------------------------------------------
__global__ void proj_wmma_kernel(const float* __restrict__ X,
                                 const float* __restrict__ W,
                                 const float* __restrict__ bias,
                                 void* __restrict__ out,
                                 int rows, int posN, int mode, float scale) {
  const int lane = threadIdx.x & 31;
  const int half = lane >> 4;
  const int lidx = lane & 15;
  const int wid  = blockIdx.x * (blockDim.x >> 5) + (threadIdx.x >> 5);
  const int tiles_n = DIMD / 16;
  const int tile_m  = wid / tiles_n;
  const int tile_n  = wid % tiles_n;
  if (tile_m * 16 >= rows) return;

  const float* __restrict__ arow = X + (size_t)(tile_m * 16 + lidx) * DIMD;
  const float* __restrict__ wrow = W + (size_t)(tile_n * 16 + lidx) * DIMD;

  v8f acc = {};
#pragma unroll
  for (int kt = 0; kt < DIMD / 32; ++kt) {
    const int k0 = kt * 32;
    v16bf A, Bm;
#pragma unroll
    for (int i = 0; i < 8; ++i) {            // A 16x32 bf16 lane layout
      A[i]     = (__bf16)arow[k0 + half * 16 + i];
      A[8 + i] = (__bf16)arow[k0 + half * 16 + 8 + i];
    }
#pragma unroll
    for (int e = 0; e < 16; ++e)             // B 32x16: K = half*16 + e
      Bm[e] = (__bf16)wrow[k0 + half * 16 + e];
    acc = wmma_bf16(A, Bm, acc);
  }

  const float bv = bias[tile_n * 16 + lidx];
#pragma unroll
  for (int j = 0; j < 8; ++j) acc[j] = (acc[j] + bv) * scale;

#pragma unroll
  for (int j = 0; j < 8; ++j) {              // C: row = half*8+j, col = lidx
    const int r = tile_m * 16 + half * 8 + j;
    const int c = tile_n * 16 + lidx;
    const float v = acc[j];
    if (mode == 3) {
      ((float*)out)[(size_t)r * DIMD + c] = v;
    } else {
      const int b = r / posN, pos = r % posN;
      const int h = c >> 5, hc = c & 31;
      if (mode == 2)
        ((__bf16*)out)[((size_t)(b * NHEAD + h) * HD + hc) * posN + pos] = (__bf16)v;
      else
        ((__bf16*)out)[((size_t)(b * NHEAD + h) * posN + pos) * HD + hc] = (__bf16)v;
    }
  }
}

// ---------------------------------------------------------------------------
// Flash attention. Block = 8 waves = 128 queries of one (b,h).
// Per 64-key chunk: workgroup async-stages K (64x32) and V (32x64) bf16 into
// LDS once; each wave then runs 4 score WMMAs + online softmax + 4 ctx WMMAs.
// ---------------------------------------------------------------------------
__global__ void flash_attn_kernel(const __bf16* __restrict__ qbf,
                                  const __bf16* __restrict__ kbf,
                                  const __bf16* __restrict__ vbfT,
                                  const float* __restrict__ amask,
                                  float* __restrict__ ctxf) {
  __shared__ __align__(16) __bf16 kstage[2][CHUNK * HD];   // (key, hd)
  __shared__ __align__(16) __bf16 vstage[2][HD * CHUNK];   // (hd, key)
  __shared__ __align__(16) __bf16 ptile[8][16 * 72];       // per-wave P relayout

  const int tid  = threadIdx.x;
  const int lane = tid & 31;
  const int half = lane >> 4;
  const int lidx = lane & 15;
  const int wv   = tid >> 5;
  const int bh   = blockIdx.x >> 3;            // 16 (b,h) pairs
  const int qblk = blockIdx.x & 7;             // 8 query blocks of 128
  const int qt   = qblk * 8 + wv;              // this wave's 16-query tile
  const int b    = bh >> 3;
  const int h    = bh & 7;

  const __bf16* kbase = kbf  + (size_t)bh * SEQ * HD;
  const __bf16* vbase = vbfT + (size_t)bh * HD * SEQ;
  const float*  mrow  = amask + (size_t)b * SEQ;
  __bf16* myp = ptile[wv];

  // per-thread staging assignment: one b128 of K + one b128 of V per chunk
  const int kkey = tid >> 2, kpart = (tid & 3) * 8;   // K: 64 keys x 32 hd
  const int vn   = tid >> 3, vpart = (tid & 7) * 8;   // V: 32 hd x 64 keys

  // Q tile in A layout
  const __bf16* qp = qbf + ((size_t)bh * NQTOK + qt * 16 + lidx) * HD;
  v16bf qa;
#pragma unroll
  for (int i = 0; i < 8; ++i) {
    qa[i]     = qp[half * 16 + i];
    qa[8 + i] = qp[half * 16 + 8 + i];
  }

  float rmax[8], rsum[8];
#pragma unroll
  for (int j = 0; j < 8; ++j) { rmax[j] = -1e30f; rsum[j] = 0.0f; }
  v8f ctx0 = {}, ctx1 = {};

  const int NIT = SEQ / CHUNK;                 // 128 iterations

  // prologue: stage chunk 0 into buffer 0
  async_copy_b128(kbase + (size_t)kkey * HD + kpart, &kstage[0][kkey * HD + kpart]);
  async_copy_b128(vbase + (size_t)vn * SEQ + vpart,  &vstage[0][vn * CHUNK + vpart]);

  for (int it = 0; it < NIT; ++it) {
    const int cur = it & 1;
    const int s0  = it * CHUNK;
    if (it + 1 < NIT) {                        // issue next chunk, then wait
      const int s1 = s0 + CHUNK;               // for the current one only
      async_copy_b128(kbase + (size_t)kkey * HD + s1 * HD + kpart,
                      &kstage[cur ^ 1][kkey * HD + kpart]);
      async_copy_b128(vbase + (size_t)vn * SEQ + s1 + vpart,
                      &vstage[cur ^ 1][vn * CHUNK + vpart]);
      asm volatile("s_wait_asynccnt 0x2" ::: "memory");
    } else {
      asm volatile("s_wait_asynccnt 0x0" ::: "memory");
    }
    __syncthreads();                           // staged chunk visible to WG

    // ---- scores: 4 x WMMA (16 queries x 64 keys) ----
    v8f sc[4];
#pragma unroll
    for (int c = 0; c < 4; ++c) {
      v16bf kb = *(const v16bf*)(&kstage[cur][(c * 16 + lidx) * HD + half * 16]);
      v8f z = {};
      sc[c] = wmma_bf16(qa, kb, z);            // scale folded into q
      const float m = mrow[s0 + c * 16 + lidx] * -100.0f;
#pragma unroll
      for (int j = 0; j < 8; ++j) sc[c][j] += m;
    }

    // ---- online softmax across the 64-key chunk ----
#pragma unroll
    for (int j = 0; j < 8; ++j) {
      float cm = fmaxf(fmaxf(sc[0][j], sc[1][j]), fmaxf(sc[2][j], sc[3][j]));
#pragma unroll
      for (int msk = 1; msk < 16; msk <<= 1)
        cm = fmaxf(cm, __shfl_xor(cm, msk, 32));
      const float nmax  = fmaxf(rmax[j], cm);
      const float alpha = __expf(rmax[j] - nmax);
      float ps = 0.0f;
#pragma unroll
      for (int c = 0; c < 4; ++c) {
        const float p = __expf(sc[c][j] - nmax);
        ps += p;
        myp[(half * 8 + j) * 72 + c * 16 + lidx] = (__bf16)p;  // C -> LDS
      }
#pragma unroll
      for (int msk = 1; msk < 16; msk <<= 1)
        ps += __shfl_xor(ps, msk, 32);
      rsum[j] = rsum[j] * alpha + ps;
      rmax[j] = nmax;
      ctx0[j] *= alpha;
      ctx1[j] *= alpha;
    }
    asm volatile("s_wait_dscnt 0x0" ::: "memory");   // P tile relayout fence

    // ---- P in A layout: two 16x32 tiles (keys 0-31, 32-63) ----
    v16bf pa0, pa1;
#pragma unroll
    for (int i = 0; i < 8; ++i) {
      pa0[i]     = myp[lidx * 72 + half * 16 + i];
      pa0[8 + i] = myp[lidx * 72 + half * 16 + 8 + i];
      pa1[i]     = myp[lidx * 72 + 32 + half * 16 + i];
      pa1[8 + i] = myp[lidx * 72 + 32 + half * 16 + 8 + i];
    }

    // ---- ctx accumulate: 4 x WMMA (P x V), V from LDS (hd, key) ----
#pragma unroll
    for (int nh = 0; nh < 2; ++nh) {
      v16bf vb0 = *(const v16bf*)(&vstage[cur][(nh * 16 + lidx) * CHUNK + half * 16]);
      v16bf vb1 = *(const v16bf*)(&vstage[cur][(nh * 16 + lidx) * CHUNK + 32 + half * 16]);
      if (nh == 0) { ctx0 = wmma_bf16(pa0, vb0, ctx0); ctx0 = wmma_bf16(pa1, vb1, ctx0); }
      else         { ctx1 = wmma_bf16(pa0, vb0, ctx1); ctx1 = wmma_bf16(pa1, vb1, ctx1); }
    }
    __syncthreads();                           // done reading cur buffer
  }

  // normalize, write merged-head context (B, NQ, 256) f32
#pragma unroll
  for (int j = 0; j < 8; ++j) {
    const float inv = 1.0f / rsum[j];
    const int qrow = qt * 16 + half * 8 + j;
    float* op = ctxf + ((size_t)(b * NQTOK + qrow)) * DIMD + h * HD;
    op[lidx]      = ctx0[j] * inv;
    op[16 + lidx] = ctx1[j] * inv;
  }
}

// ---------------------------------------------------------------------------
extern "C" void kernel_launch(void* const* d_in, const int* in_sizes, int n_in,
                              void* d_out, int out_size, void* d_ws, size_t ws_size,
                              hipStream_t stream) {
  const float* q_in = (const float*)d_in[0];
  const float* k_in = (const float*)d_in[1];
  const float* v_in = (const float*)d_in[2];
  const float* am   = (const float*)d_in[3];
  const float* Wq   = (const float*)d_in[4];
  const float* bq   = (const float*)d_in[5];
  const float* Wk   = (const float*)d_in[6];
  const float* bk   = (const float*)d_in[7];
  const float* Wv   = (const float*)d_in[8];
  const float* bv   = (const float*)d_in[9];
  const float* Wo   = (const float*)d_in[10];
  const float* bo   = (const float*)d_in[11];

  char* ws = (char*)d_ws;
  size_t off = 0;
  __bf16* qbf  = (__bf16*)(ws + off); off += (size_t)BATCH * NHEAD * NQTOK * HD * 2;
  __bf16* kbf  = (__bf16*)(ws + off); off += (size_t)BATCH * NHEAD * SEQ   * HD * 2;
  __bf16* vbfT = (__bf16*)(ws + off); off += (size_t)BATCH * NHEAD * SEQ   * HD * 2;
  float*  ctxf = (float*)(ws + off);

  const float scaleq = 1.0f / sqrtf((float)HD);
  const dim3 blk(256);
  const int rows_q = BATCH * NQTOK;                     // 2048
  const int rows_s = BATCH * SEQ;                       // 16384
  const int grid_q = (rows_q / 16) * (DIMD / 16) / 8;   // 256
  const int grid_s = (rows_s / 16) * (DIMD / 16) / 8;   // 2048

  proj_wmma_kernel<<<grid_q, blk, 0, stream>>>(q_in, Wq, bq, qbf,  rows_q, NQTOK, 0, scaleq);
  proj_wmma_kernel<<<grid_s, blk, 0, stream>>>(k_in, Wk, bk, kbf,  rows_s, SEQ,   1, 1.0f);
  proj_wmma_kernel<<<grid_s, blk, 0, stream>>>(v_in, Wv, bv, vbfT, rows_s, SEQ,   2, 1.0f);

  flash_attn_kernel<<<BATCH * NHEAD * (NQTOK / 128), blk, 0, stream>>>(
      qbf, kbf, vbfT, am, ctxf);

  proj_wmma_kernel<<<grid_q, blk, 0, stream>>>(ctxf, Wo, bo, d_out, rows_q, NQTOK, 3, 1.0f);
}
